// net_91096256348626
// MI455X (gfx1250) — matrix operand order repaired and checked by reading, not compiled, not run on hardware
//
#include <hip/hip_runtime.h>

typedef __attribute__((ext_vector_type(16))) _Float16 v16h;
typedef __attribute__((ext_vector_type(8)))  _Float16 v8h;
typedef __attribute__((ext_vector_type(8)))  float    v8f;

// ---------------- zero fill ----------------
__global__ void zero_f32(float* __restrict__ p, long n) {
    long i = (long)blockIdx.x * blockDim.x + threadIdx.x;
    long stride = (long)gridDim.x * blockDim.x;
    for (; i < n; i += stride) p[i] = 0.0f;
}

// ---------------- kernel-weighted message passing (K=2), thread per (edge,channel) ----------------
// out[dst, offOut + j*Cin + c] += kv[j*E + e] * hin[src, offIn + c]
__global__ void aniso_conv(const float* __restrict__ hin, int strideIn, int offIn, int Cin,
                           const float* __restrict__ kv,
                           const int* __restrict__ src, const int* __restrict__ dst,
                           float* __restrict__ out, int strideOut, int offOut,
                           long total, int E) {
    long t = (long)blockIdx.x * blockDim.x + threadIdx.x;
    if (t >= total) return;
    int e = (int)(t / Cin);
    int c = (int)(t % Cin);
    int s = src[e];
    int d = dst[e];
    float m = hin[(long)s * strideIn + offIn + c];
    float* __restrict__ orow = out + (long)d * strideOut + offOut;
    atomicAdd(&orow[c],       kv[e]     * m);
    atomicAdd(&orow[Cin + c], kv[E + e] * m);
}

// ---------------- pack fp32 activations -> zero-padded f16 [n_pad][KP] ----------------
__global__ void pack_a(const float* __restrict__ A, _Float16* __restrict__ out,
                       int n, int CIN, int KP, long total) {
    long i = (long)blockIdx.x * blockDim.x + threadIdx.x;
    long stride = (long)gridDim.x * blockDim.x;
    for (; i < total; i += stride) {
        long row = i / KP;
        int  k   = (int)(i % KP);
        float v  = (row < n && k < CIN) ? A[row * (long)CIN + k] : 0.0f;
        out[i] = (_Float16)v;
    }
}

// ---------------- pack fp32 weights [CIN][COUT] -> transposed padded f16 [COUTP][KP] ----------------
__global__ void pack_w(const float* __restrict__ W, _Float16* __restrict__ out,
                       int CIN, int COUT, int KP, int COUTP) {
    long total = (long)COUTP * KP;
    long i = (long)blockIdx.x * blockDim.x + threadIdx.x;
    long stride = (long)gridDim.x * blockDim.x;
    for (; i < total; i += stride) {
        int c = (int)(i / KP);
        int k = (int)(i % KP);
        float v = (c < COUT && k < CIN) ? W[(long)k * COUT + c] : 0.0f;
        out[i] = (_Float16)v;
    }
}

// ---------------- WMMA GEMM: out[n,COUT] = relu?(A @ W + bias) ----------------
// Af: f16 [n_pad][KP] zero-padded.  Wt: f16 [COUTP][KP] transposed, zero-padded.
// Block = (COUTP/16) waves; wave w owns column tile w; grid.x = row tiles.
// Branch-free inner loop: A fragment = 2 aligned b128 loads, B fragment = 1 aligned 32B load.
// Epilogue: single tile-level predicate, then 8 unconditional (clause-able) stores.
template <int KP, int COUT, int COUTP, bool RELU>
__global__ void gemm_wmma(const _Float16* __restrict__ Af, const _Float16* __restrict__ Wt,
                          const float* __restrict__ bias, float* __restrict__ out, int n) {
    const int lane = threadIdx.x & 31;
    const int wave = threadIdx.x >> 5;
    const int half = lane >> 4;
    const int lx   = lane & 15;
    const int row0 = blockIdx.x * 16;

    const _Float16* __restrict__ arow = Af + (size_t)(row0 + lx) * KP + half * 16;
    const _Float16* __restrict__ wrow = Wt + (size_t)(wave * 16 + lx) * KP + half * 16;
    __builtin_prefetch(wrow, 0, 0);

    v8f acc = {};
#pragma unroll
    for (int kb = 0; kb < KP; kb += 32) {
        v8h alo = *(const v8h*)(arow + kb);        // K = half*16 + 0..7
        v8h ahi = *(const v8h*)(arow + kb + 8);    // K = half*16 + 8..15
        v16h a  = __builtin_shufflevector(alo, ahi,
                    0,1,2,3,4,5,6,7, 8,9,10,11,12,13,14,15);
        v16h b  = *(const v16h*)(wrow + kb);       // K = half*16 + 0..15, contiguous
        acc = __builtin_amdgcn_wmma_f32_16x16x32_f16(
            /*neg_a=*/false, a, /*neg_b=*/false, b,
            /*c_mod=*/(short)0, acc, /*reuse_a=*/false, /*reuse_b=*/false);
    }

    const int  cN       = wave * 16 + lx;
    const bool colOK    = (COUTP == COUT) || (cN < COUT);          // only lin0 tile 4 can fail
    const bool fullTile = (row0 + 16 <= n);
    if (colOK) {
        const float bv = bias[cN];
        float* __restrict__ ocol = out + (size_t)(row0 + half * 8) * COUT + cN;
        if (fullTile) {
            // hot path: no per-row predication -> stores can issue as a clause
#pragma unroll
            for (int r = 0; r < 8; ++r) {
                float v = acc[r] + bv;
                if (RELU) v = fmaxf(v, 0.0f);
                ocol[(size_t)r * COUT] = v;
            }
        } else {
#pragma unroll
            for (int r = 0; r < 8; ++r) {
                int row = row0 + half * 8 + r;
                if (row < n) {
                    float v = acc[r] + bv;
                    if (RELU) v = fmaxf(v, 0.0f);
                    ocol[(size_t)r * COUT] = v;
                }
            }
        }
    }
}

// ---------------- L2 row normalize (64 channels, one wave per row) ----------------
__global__ void l2norm64(const float* __restrict__ in, float* __restrict__ out, int n) {
    int row  = blockIdx.x * (blockDim.x / 32) + (threadIdx.x / 32);
    int lane = threadIdx.x & 31;
    if (row >= n) return;
    float a = in[(long)row * 64 + lane];
    float b = in[(long)row * 64 + 32 + lane];
    float s = a * a + b * b;
#pragma unroll
    for (int off = 16; off > 0; off >>= 1) s += __shfl_xor(s, off, 32);
    float inv = 1.0f / fmaxf(sqrtf(s), 1e-12f);
    out[(long)row * 64 + lane]      = a * inv;
    out[(long)row * 64 + 32 + lane] = b * inv;
}

extern "C" void kernel_launch(void* const* d_in, const int* in_sizes, int n_in,
                              void* d_out, int out_size, void* d_ws, size_t ws_size,
                              hipStream_t stream) {
    (void)n_in; (void)out_size; (void)ws_size;

    const float* x      = (const float*)d_in[0];   // [N,3]
    const float* kdd    = (const float*)d_in[1];   // [2,E]
    const float* kda    = (const float*)d_in[2];   // [2,E]
    const float* lin0_w = (const float*)d_in[3];   // [36,72]
    const float* lin0_b = (const float*)d_in[4];   // [72]
    const float* mlp_w1 = (const float*)d_in[5];   // [144,128]
    const float* mlp_b1 = (const float*)d_in[6];   // [128]
    const float* mlp_w2 = (const float*)d_in[7];   // [128,64]
    const float* mlp_b2 = (const float*)d_in[8];   // [64]
    const int*   src    = (const int*)d_in[9];     // [E]
    const int*   dst    = (const int*)d_in[10];    // [E]
    float*       outp   = (float*)d_out;           // [N,64]

    const int N = in_sizes[0] / 3;
    const int E = in_sizes[9];
    const int rowT  = (N + 15) / 16;
    const int n_pad = rowT * 16;

    // --- workspace: three rotating fp32/f16 regions + packed weights ---
    char* ws = (char*)d_ws;
    size_t s0 = (((size_t)N * 128 * 4) + 255) & ~(size_t)255;     // hcat / lin / mlp1
    size_t s1 = (((size_t)N * 144 * 4) + 255) & ~(size_t)255;     // da1 / da2 / pm1
    size_t s2 = (((size_t)n_pad * 160 * 2) + 255) & ~(size_t)255; // pa1 / pa2 / mlp2
    char* R0 = ws;
    char* R1 = ws + s0;
    char* R2 = ws + s0 + s1;
    char* RW = ws + s0 + s1 + s2;

    float*     hcat = (float*)R0;                 // [N,18]
    float*     lin  = (float*)R0;                 // [N,72]   (after hcat dead)
    float*     mlp1 = (float*)R0;                 // [N,128]  (after lin dead)
    float*     da1  = (float*)R1;                 // [N,36]
    float*     da2  = (float*)R1;                 // [N,144]  (after da1 dead)
    _Float16*  pm1  = (_Float16*)R1;              // [n_pad,128] (after da2 dead)
    _Float16*  pa1  = (_Float16*)R2;              // [n_pad,64]
    _Float16*  pa2  = (_Float16*)R2;              // [n_pad,160] (after pa1 dead)
    float*     mlp2 = (float*)R2;                 // [N,64]      (after pa2 dead)
    _Float16*  wt0  = (_Float16*)RW;              // [80,64]
    _Float16*  wt1  = wt0 + 80 * 64;              // [128,160]
    _Float16*  wt2  = wt1 + 128 * 160;            // [64,128]

    const int ZB = 2048, PB = 4096;
    auto cgrid = [](long total) { return dim3((unsigned)((total + 255) / 256)); };

    // --- DD phase: x -> hcat[:,0:6] -> hcat[:,6:18] ---
    zero_f32<<<ZB, 256, 0, stream>>>(hcat, (long)N * 18);
    aniso_conv<<<cgrid((long)E * 3), 256, 0, stream>>>(x,    3, 0,  3, kdd, src, dst, hcat, 18, 0, (long)E * 3, E);
    aniso_conv<<<cgrid((long)E * 6), 256, 0, stream>>>(hcat, 18, 0, 6, kdd, src, dst, hcat, 18, 6, (long)E * 6, E);

    // --- DA hop 1: hcat[:,0:18] -> da1[:,0:36] ---
    zero_f32<<<ZB, 256, 0, stream>>>(da1, (long)N * 36);
    aniso_conv<<<cgrid((long)E * 18), 256, 0, stream>>>(hcat, 18, 0, 18, kda, src, dst, da1, 36, 0, (long)E * 18, E);

    // --- lin0 + ReLU: [N,36] @ [36,72] via WMMA ---
    pack_a<<<PB, 256, 0, stream>>>(da1, pa1, N, 36, 64, (long)n_pad * 64);
    pack_w<<<32, 256, 0, stream>>>(lin0_w, wt0, 36, 72, 64, 80);
    gemm_wmma<64, 72, 80, true><<<rowT, 160, 0, stream>>>(pa1, wt0, lin0_b, lin, N);

    // --- DA hop 2: lin[:,0:72] -> da2[:,0:144] ---
    zero_f32<<<ZB, 256, 0, stream>>>(da2, (long)N * 144);
    aniso_conv<<<cgrid((long)E * 72), 256, 0, stream>>>(lin, 72, 0, 72, kda, src, dst, da2, 144, 0, (long)E * 72, E);

    // --- MLP1 + ReLU: [N,144] @ [144,128] via WMMA ---
    pack_a<<<PB, 256, 0, stream>>>(da2, pa2, N, 144, 160, (long)n_pad * 160);
    pack_w<<<96, 256, 0, stream>>>(mlp_w1, wt1, 144, 128, 160, 128);
    gemm_wmma<160, 128, 128, true><<<rowT, 256, 0, stream>>>(pa2, wt1, mlp_b1, mlp1, N);

    // --- MLP2: [N,128] @ [128,64] via WMMA ---
    pack_a<<<PB, 256, 0, stream>>>(mlp1, pm1, N, 128, 128, (long)n_pad * 128);
    pack_w<<<48, 256, 0, stream>>>(mlp_w2, wt2, 128, 64, 128, 64);
    gemm_wmma<128, 64, 64, false><<<rowT, 128, 0, stream>>>(pm1, wt2, mlp_b2, mlp2, N);

    // --- L2 normalize rows of 64 ---
    l2norm64<<<(N + 7) / 8, 256, 0, stream>>>(mlp2, outp, N);
}
